// GraphIsomorphismLayer_68461778698670
// MI455X (gfx1250) — matrix operand order repaired and checked by reading, not compiled, not run on hardware
//
#include <hip/hip_runtime.h>

// Problem constants (from reference)
#define Bc 8
#define Nc 16384
#define Ec 32768
#define Dc 256
#define Hc 512

typedef __attribute__((ext_vector_type(16))) _Float16 v16h;
typedef __attribute__((ext_vector_type(8)))  _Float16 v8h;
typedef __attribute__((ext_vector_type(8)))  float    v8f;

#define PAD_D 264   // 256 + 8 halves: row stride 528B = 33*16 -> aligned + bank-spread
#define PAD_H 520   // 512 + 8 halves: row stride 1040B = 65*16

// ---------------------------------------------------------------------------
// Kernel 1: zero the message scratch (B*N*D floats = 128 MB) with b128 stores
// ---------------------------------------------------------------------------
__global__ void gin_zero_ws(float4* __restrict__ ws) {
    long i = (long)blockIdx.x * blockDim.x + threadIdx.x;
    ws[i] = make_float4(0.f, 0.f, 0.f, 0.f);
}

// ---------------------------------------------------------------------------
// Kernel 2: per-edge gather + bond-embedding add + atomic scatter to receiver
// ---------------------------------------------------------------------------
__global__ void gin_edge_scatter(const float* __restrict__ nodes,
                                 const int*   __restrict__ edge_feats,
                                 const int*   __restrict__ senders,
                                 const int*   __restrict__ receivers,
                                 const float* __restrict__ edge_emb,
                                 float*       __restrict__ message) {
    const int e = blockIdx.x;
    const int b = blockIdx.y;
    const int d = threadIdx.x;
    const long be = (long)b * Ec + e;
    const int s  = senders[be];
    const int r  = receivers[be];
    const int f0 = edge_feats[be * 3 + 0];
    const int f1 = edge_feats[be * 3 + 1];
    const int f2 = edge_feats[be * 3 + 2];

    float v = nodes[((long)b * Nc + s) * Dc + d]
            + edge_emb[((long)(0 * 8 + f0)) * Dc + d]
            + edge_emb[((long)(1 * 8 + f1)) * Dc + d]
            + edge_emb[((long)(2 * 8 + f2)) * Dc + d];

    unsafeAtomicAdd(&message[((long)b * Nc + r) * Dc + d], v);   // global_atomic_add_f32
}

// ---------------------------------------------------------------------------
// Kernels 3a/3b: convert weights to split-f16 (hi/lo), transposed column-major
// so B-fragments become contiguous 32-byte per-lane loads.
//  w1 (D x H) -> w1t_{hi,lo}[n][k] : n<H, k<D
//  w2 (H x D) -> w2t_{hi,lo}[n][k] : n<D, k<H
// ---------------------------------------------------------------------------
__global__ void gin_prep_w1(const float* __restrict__ w1,
                            _Float16* __restrict__ hi, _Float16* __restrict__ lo) {
    const int n = blockIdx.x;      // 0..511
    const int k = threadIdx.x;     // 0..255
    const float v = w1[(long)k * Hc + n];
    const _Float16 h = (_Float16)v;
    hi[(long)n * Dc + k] = h;
    lo[(long)n * Dc + k] = (_Float16)(v - (float)h);
}

__global__ void gin_prep_w2(const float* __restrict__ w2,
                            _Float16* __restrict__ hi, _Float16* __restrict__ lo) {
    const int n = blockIdx.x;      // 0..255
    const int k = threadIdx.x;     // 0..511
    const float v = w2[(long)k * Dc + n];
    const _Float16 h = (_Float16)v;
    hi[(long)n * Hc + k] = h;
    lo[(long)n * Hc + k] = (_Float16)(v - (float)h);
}

// ---------------------------------------------------------------------------
// Kernel 4: fused GIN MLP, split-f16 compensated GEMM on V_WMMA_F32_16X16X32_F16
// grid = (N/16, B), 256 threads = 8 wave32s, one 16-node tile per block.
// D = A_hi*B_hi + A_hi*B_lo + A_lo*B_hi  (fp32 accum; lo*lo ~2^-22, dropped)
// ---------------------------------------------------------------------------
__device__ __forceinline__ v16h load_afrag(const _Float16* p) {
    // A fragment in this code's contiguous k order: this lane loads K = base..base+15 (matches B)
    v8h p0 = *(const v8h*)(p);
    v8h p1 = *(const v8h*)(p + 8);
    return __builtin_shufflevector(p0, p1, 0,1,2,3,4,5,6,7,8,9,10,11,12,13,14,15);
}

__launch_bounds__(256)
__global__ void gin_mlp(const float* __restrict__ nodes,
                        const float* __restrict__ message,
                        const _Float16* __restrict__ w1t_hi,
                        const _Float16* __restrict__ w1t_lo,
                        const float* __restrict__ b1,
                        const _Float16* __restrict__ w2t_hi,
                        const _Float16* __restrict__ w2t_lo,
                        const float* __restrict__ b2,
                        const float* __restrict__ eps,
                        float*       __restrict__ out) {
    __shared__ _Float16 shA_hi[16][PAD_D];
    __shared__ _Float16 shA_lo[16][PAD_D];
    __shared__ _Float16 shH_hi[16][PAD_H];
    __shared__ _Float16 shH_lo[16][PAD_H];

    const int b    = blockIdx.y;
    const int row0 = blockIdx.x * 16;
    const int tid  = threadIdx.x;
    const float scale = 1.0f + eps[0];

    // stage h = (1+eps)*x + msg, split into f16 hi/lo (coalesced fp32 reads)
    #pragma unroll 4
    for (int i = 0; i < 16; ++i) {
        const long g = ((long)b * Nc + row0 + i) * Dc + tid;
        const float v = scale * nodes[g] + message[g];
        const _Float16 h = (_Float16)v;
        shA_hi[i][tid] = h;
        shA_lo[i][tid] = (_Float16)(v - (float)h);
    }
    __syncthreads();

    const int wave  = tid >> 5;
    const int lane  = tid & 31;
    const int col   = lane & 15;
    const int half  = lane >> 4;       // 0: low K-half lanes, 1: high
    const int rbase = half * 8;        // C/D rows: upper lane half holds M=8..15

    // ---------------- GEMM1: h(16x256) @ W1 + b1, relu -> shH --------------
    v8f acc[4];
    #pragma unroll
    for (int t = 0; t < 4; ++t) {
        const float bias = b1[wave * 64 + t * 16 + col];
        #pragma unroll
        for (int j = 0; j < 8; ++j) acc[t][j] = bias;
    }
    for (int k0 = 0; k0 < Dc; k0 += 32) {
        const int abase = k0 + half * 16;
        const v16h a_hi = load_afrag(&shA_hi[col][abase]);
        const v16h a_lo = load_afrag(&shA_lo[col][abase]);
        const int bk = k0 + half * 16;
        #pragma unroll
        for (int t = 0; t < 4; ++t) {
            const long nrow = (long)(wave * 64 + t * 16 + col);
            const v16h b_hi = *(const v16h*)&w1t_hi[nrow * Dc + bk];
            const v16h b_lo = *(const v16h*)&w1t_lo[nrow * Dc + bk];
            acc[t] = __builtin_amdgcn_wmma_f32_16x16x32_f16(
                         false, a_hi, false, b_hi, (short)0, acc[t], false, false);
            acc[t] = __builtin_amdgcn_wmma_f32_16x16x32_f16(
                         false, a_hi, false, b_lo, (short)0, acc[t], false, false);
            acc[t] = __builtin_amdgcn_wmma_f32_16x16x32_f16(
                         false, a_lo, false, b_hi, (short)0, acc[t], false, false);
        }
    }
    #pragma unroll
    for (int t = 0; t < 4; ++t) {
        const int n0 = wave * 64 + t * 16;
        #pragma unroll
        for (int j = 0; j < 8; ++j) {
            float v = acc[t][j];
            v = v > 0.f ? v : 0.f;                  // relu
            const _Float16 h = (_Float16)v;
            shH_hi[rbase + j][n0 + col] = h;
            shH_lo[rbase + j][n0 + col] = (_Float16)(v - (float)h);
        }
    }
    __syncthreads();

    // ---------------- GEMM2: h1(16x512) @ W2 + b2 -> out -------------------
    v8f acc2[2];
    #pragma unroll
    for (int t = 0; t < 2; ++t) {
        const float bias = b2[wave * 32 + t * 16 + col];
        #pragma unroll
        for (int j = 0; j < 8; ++j) acc2[t][j] = bias;
    }
    for (int k0 = 0; k0 < Hc; k0 += 32) {
        const int abase = k0 + half * 16;
        const v16h a_hi = load_afrag(&shH_hi[col][abase]);
        const v16h a_lo = load_afrag(&shH_lo[col][abase]);
        const int bk = k0 + half * 16;
        #pragma unroll
        for (int t = 0; t < 2; ++t) {
            const long nrow = (long)(wave * 32 + t * 16 + col);
            const v16h b_hi = *(const v16h*)&w2t_hi[nrow * Hc + bk];
            const v16h b_lo = *(const v16h*)&w2t_lo[nrow * Hc + bk];
            acc2[t] = __builtin_amdgcn_wmma_f32_16x16x32_f16(
                          false, a_hi, false, b_hi, (short)0, acc2[t], false, false);
            acc2[t] = __builtin_amdgcn_wmma_f32_16x16x32_f16(
                          false, a_hi, false, b_lo, (short)0, acc2[t], false, false);
            acc2[t] = __builtin_amdgcn_wmma_f32_16x16x32_f16(
                          false, a_lo, false, b_hi, (short)0, acc2[t], false, false);
        }
    }
    #pragma unroll
    for (int t = 0; t < 2; ++t) {
        const int n0 = wave * 32 + t * 16;
        #pragma unroll
        for (int j = 0; j < 8; ++j) {
            const long g = ((long)b * Nc + row0 + rbase + j) * Dc + n0 + col;
            out[g] = acc2[t][j];
        }
    }
}

// ---------------------------------------------------------------------------
extern "C" void kernel_launch(void* const* d_in, const int* in_sizes, int n_in,
                              void* d_out, int out_size, void* d_ws, size_t ws_size,
                              hipStream_t stream) {
    const float* nodes      = (const float*)d_in[0];
    const int*   edge_feats = (const int*)  d_in[1];
    const int*   senders    = (const int*)  d_in[2];
    const int*   receivers  = (const int*)  d_in[3];
    const float* edge_emb   = (const float*)d_in[4];
    const float* w1         = (const float*)d_in[5];
    const float* b1         = (const float*)d_in[6];
    const float* w2         = (const float*)d_in[7];
    const float* b2         = (const float*)d_in[8];
    const float* eps        = (const float*)d_in[9];
    float*       out        = (float*)d_out;

    // workspace layout:
    //   [0, 128MB)      message  (B*N*D fp32)
    //   then 4 x 256KB  w1t_hi, w1t_lo, w2t_hi, w2t_lo (split-f16 weights)
    float*    message = (float*)d_ws;
    _Float16* w1t_hi  = (_Float16*)((char*)d_ws + (size_t)Bc * Nc * Dc * 4);
    _Float16* w1t_lo  = w1t_hi + (size_t)Dc * Hc;
    _Float16* w2t_hi  = w1t_lo + (size_t)Dc * Hc;
    _Float16* w2t_lo  = w2t_hi + (size_t)Dc * Hc;

    // 1) zero message buffer: 33,554,432 floats = 8,388,608 float4
    gin_zero_ws<<<32768, 256, 0, stream>>>((float4*)message);

    // 2) split-f16 weight transform (transposed) — 1 MB total, negligible
    gin_prep_w1<<<Hc, Dc, 0, stream>>>(w1, w1t_hi, w1t_lo);
    gin_prep_w2<<<Dc, Hc, 0, stream>>>(w2, w2t_hi, w2t_lo);

    // 3) edge gather + scatter-add
    dim3 eg(Ec, Bc);
    gin_edge_scatter<<<eg, Dc, 0, stream>>>(nodes, edge_feats, senders,
                                            receivers, edge_emb, message);

    // 4) fused MLP with split-f16 compensated WMMA
    dim3 mg(Nc / 16, Bc);
    gin_mlp<<<mg, 256, 0, stream>>>(nodes, message,
                                    w1t_hi, w1t_lo, b1,
                                    w2t_hi, w2t_lo, b2, eps, out);
}
